// atten_module_75299366633566
// MI455X (gfx1250) — matrix operand order repaired and checked by reading, not compiled, not run on hardware
//
#include <hip/hip_runtime.h>
#include <hip/hip_bf16.h>

// ---------------------------------------------------------------------------
// Transformer-XL relative attention block for gfx1250 (MI455X).
// All GEMM-shaped math runs on v_wmma_f32_16x16x32_bf16 (bf16 A/B, f32 acc).
// Global->LDS tile staging uses CDNA5 async-DMA (global_load_async_to_lds_b128
// + s_wait_asynccnt), double-buffered so the DMA for tile k+1 overlaps the
// WMMA work on tile k.  Softmax row reductions use v_permlane16_b32 (VALU).
// BD term: since r is a broadcast [1,D] vector, BD is j-constant pre-shift;
// after _rel_shift: BD'[i,j] = c[i] (j<=i), 0 (j==i+1), c[i+1] (j>i+1).
// ---------------------------------------------------------------------------

#define S_LEN 2048
#define BATCH 2
#define DMODEL 1024
#define NHEAD 16
#define DHEAD 64
#define MROWS (S_LEN * BATCH)   // 4096
#define N3 (3 * DMODEL)         // 3072

typedef __bf16 bf16_t;
typedef __attribute__((ext_vector_type(16))) __bf16 v16bf;
typedef __attribute__((ext_vector_type(8)))  float  v8f;

// CDNA5 async global->LDS copy (16B per lane), tracked by ASYNCcnt.
__device__ __forceinline__ void async_copy_b128(void* lds_ptr, const void* gptr) {
  unsigned lds_addr = (unsigned)(unsigned long long)lds_ptr;
  unsigned long long gaddr = (unsigned long long)gptr;
  asm volatile("global_load_async_to_lds_b128 %0, %1, off"
               :: "v"(lds_addr), "v"(gaddr) : "memory");
}
__device__ __forceinline__ void wait_async() {
  asm volatile("s_wait_asynccnt 0x0" ::: "memory");
}

// 16-lane xor-shuffles on the VALU (v_permlane16_b32) instead of ds_bpermute.
#if defined(__has_builtin)
#if __has_builtin(__builtin_amdgcn_permlane16)
#define HAVE_PERMLANE16 1
#endif
#endif
__device__ __forceinline__ float lane16_xor(float v, unsigned s1, unsigned s2, int mask) {
#ifdef HAVE_PERMLANE16
  unsigned u = __float_as_uint(v);
  return __uint_as_float(__builtin_amdgcn_permlane16(u, u, s1, s2, false, false));
#else
  return __shfl_xor(v, mask, 32);
#endif
}
__device__ __forceinline__ float redmax16(float v) {
  v = fmaxf(v, lane16_xor(v, 0x67452301u, 0xEFCDAB89u, 1));
  v = fmaxf(v, lane16_xor(v, 0x54761032u, 0xDCFE98BAu, 2));
  v = fmaxf(v, lane16_xor(v, 0x32107654u, 0xBA98FEDCu, 4));
  v = fmaxf(v, lane16_xor(v, 0xFEDCBA98u, 0x76543210u, 8));
  return v;
}
__device__ __forceinline__ float redsum16(float v) {
  v += lane16_xor(v, 0x67452301u, 0xEFCDAB89u, 1);
  v += lane16_xor(v, 0x54761032u, 0xDCFE98BAu, 2);
  v += lane16_xor(v, 0x32107654u, 0xBA98FEDCu, 4);
  v += lane16_xor(v, 0xFEDCBA98u, 0x76543210u, 8);
  return v;
}

// ---------------------------------------------------------------------------
// LayerNorm: one block per row (S*B rows), D=1024
// ---------------------------------------------------------------------------
__global__ void k_ln(const float* __restrict__ x, const float* __restrict__ g,
                     const float* __restrict__ be, float* __restrict__ xn,
                     bf16_t* __restrict__ xnb) {
  __shared__ float r1[256], r2[256];
  const int row = blockIdx.x;
  const float* xr = x + (size_t)row * DMODEL;
  float s = 0.f, s2 = 0.f;
  for (int c = threadIdx.x; c < DMODEL; c += 256) {
    float v = xr[c]; s += v; s2 += v * v;
  }
  r1[threadIdx.x] = s; r2[threadIdx.x] = s2;
  __syncthreads();
  for (int st = 128; st > 0; st >>= 1) {
    if (threadIdx.x < st) {
      r1[threadIdx.x] += r1[threadIdx.x + st];
      r2[threadIdx.x] += r2[threadIdx.x + st];
    }
    __syncthreads();
  }
  const float mu  = r1[0] * (1.0f / DMODEL);
  const float var = r2[0] * (1.0f / DMODEL) - mu * mu;
  const float rstd = rsqrtf(var + 1e-5f);
  for (int c = threadIdx.x; c < DMODEL; c += 256) {
    float v = (xr[c] - mu) * rstd * g[c] + be[c];
    xn[(size_t)row * DMODEL + c]  = v;
    xnb[(size_t)row * DMODEL + c] = (bf16_t)v;
  }
}

// f32 -> bf16 bulk convert
__global__ void k_cvt(const float* __restrict__ in, bf16_t* __restrict__ out, int n) {
  int i = blockIdx.x * 256 + threadIdx.x;
  if (i < n) out[i] = (bf16_t)in[i];
}

// rvec[j] = sum_d r[d] * Wr[d][j]
__global__ void k_rvec(const float* __restrict__ r, const float* __restrict__ Wr,
                       float* __restrict__ rv) {
  int j = blockIdx.x * 256 + threadIdx.x;
  if (j >= DMODEL) return;
  float s = 0.f;
  for (int d = 0; d < DMODEL; ++d) s += r[d] * Wr[(size_t)d * DMODEL + j];
  rv[j] = s;
}

// ---------------------------------------------------------------------------
// Tiled bf16 WMMA GEMM: C[M,N] (f32) = A[M,K](bf16) * B[K,N](bf16) [+ Res]
// Block: 128 threads (4 waves); block tile 128x64; K-step 32; LDS double-
// buffered so async DMA of tile k+1 overlaps WMMA work on tile k.
// Each wave: 32 rows x 64 cols = 2x4 accumulators; B-frag reused across the
// two row strips.
// ---------------------------------------------------------------------------
template <int HAS_RES>
__global__ void __launch_bounds__(128, 1)
k_gemm(const bf16_t* __restrict__ A, const bf16_t* __restrict__ Bm,
       const float* __restrict__ Res, float* __restrict__ C,
       int M, int N, int K) {
  __shared__ __align__(16) bf16_t As[2][128][32];
  __shared__ __align__(16) bf16_t Bs[2][32][64];

  const int tid = threadIdx.x;
  const int l   = tid & 31;
  const int w   = tid >> 5;
  const int tile_m = blockIdx.y * 128;
  const int tile_n = blockIdx.x * 64;

  v8f acc[2][4] = {};

  const int rb = tid >> 2, cb = (tid & 3) * 16;   // B-tile: 32 rows x 4 groups
  const int kh   = (l >> 4) * 16;                 // A-frag K-half (contiguous)
  const int kbh  = (l & 16) ? 16 : 0;             // B-frag K-half

  auto load_tiles = [&](int k0, int buf) {
    const bf16_t* arow = A + (size_t)(tile_m + tid) * K + k0;
#pragma unroll
    for (int c = 0; c < 4; ++c)
      async_copy_b128(&As[buf][tid][c * 8], arow + c * 8);
    async_copy_b128(&Bs[buf][rb][cb],     Bm + (size_t)(k0 + rb) * N + tile_n + cb);
    async_copy_b128(&Bs[buf][rb][cb + 8], Bm + (size_t)(k0 + rb) * N + tile_n + cb + 8);
  };

  load_tiles(0, 0);
  wait_async();
  __syncthreads();

  const int nk = K / 32;
  for (int ks = 0; ks < nk; ++ks) {
    const int buf = ks & 1;
    if (ks + 1 < nk) load_tiles((ks + 1) * 32, buf ^ 1);

    v16bf af[2];
#pragma unroll
    for (int st = 0; st < 2; ++st) {
      int mrow = w * 32 + st * 16 + (l & 15);
#pragma unroll
      for (int p = 0; p < 8; ++p) {
        int kb = kh + p * 2;
        af[st][2 * p]     = As[buf][mrow][kb];
        af[st][2 * p + 1] = As[buf][mrow][kb + 1];
      }
    }
#pragma unroll
    for (int nc = 0; nc < 4; ++nc) {
      v16bf bfm;
      int ncol = nc * 16 + (l & 15);
#pragma unroll
      for (int p = 0; p < 8; ++p) {
        bfm[2 * p]     = Bs[buf][kbh + 2 * p][ncol];
        bfm[2 * p + 1] = Bs[buf][kbh + 2 * p + 1][ncol];
      }
      acc[0][nc] = __builtin_amdgcn_wmma_f32_16x16x32_bf16(
          false, af[0], false, bfm, (short)0, acc[0][nc], false, false);
      acc[1][nc] = __builtin_amdgcn_wmma_f32_16x16x32_bf16(
          false, af[1], false, bfm, (short)0, acc[1][nc], false, false);
    }
    wait_async();      // next tile's DMA (issued above) has landed
    __syncthreads();   // all waves done reading buf & done writing buf^1
  }

  const int half8 = (l & 16) ? 8 : 0;
#pragma unroll
  for (int st = 0; st < 2; ++st)
#pragma unroll
    for (int nc = 0; nc < 4; ++nc)
#pragma unroll
      for (int p = 0; p < 8; ++p) {
        int row = tile_m + w * 32 + st * 16 + half8 + p;
        int col = tile_n + nc * 16 + (l & 15);
        float v = acc[st][nc][p];
        if (HAS_RES) v += Res[(size_t)row * N + col];
        C[(size_t)row * N + col] = v;
      }
}

// ---------------------------------------------------------------------------
// Prep: split qkv, add biases, convert to bf16 in [b][h][s][d] layout,
// and compute c[i,b,h] = dot(q + rr, rvec).  64 threads per (s,b,h).
// ---------------------------------------------------------------------------
__global__ void k_prep(const float* __restrict__ qkv, const float* __restrict__ rw,
                       const float* __restrict__ rr, const float* __restrict__ rv,
                       bf16_t* __restrict__ qp, bf16_t* __restrict__ kk,
                       bf16_t* __restrict__ vv, float* __restrict__ cbd) {
  __shared__ float red[64];
  const int blk = blockIdx.x;              // (s*B + b)*H + h
  const int h = blk % NHEAD;
  const int sb = blk / NHEAD;
  const int b = sb % BATCH;
  const int s = sb / BATCH;
  const int d = threadIdx.x;

  const float* base = qkv + (size_t)sb * N3;
  const float q = base[h * DHEAD + d];
  const float k = base[DMODEL + h * DHEAD + d];
  const float v = base[2 * DMODEL + h * DHEAD + d];

  const size_t o = ((size_t)(b * NHEAD + h) * S_LEN + s) * DHEAD + d;
  qp[o] = (bf16_t)(q + rw[h * DHEAD + d]);
  kk[o] = (bf16_t)k;
  vv[o] = (bf16_t)v;

  red[d] = (q + rr[h * DHEAD + d]) * rv[h * DHEAD + d];
  __syncthreads();
  for (int st = 32; st > 0; st >>= 1) {
    if (d < st) red[d] += red[d + st];
    __syncthreads();
  }
  if (d == 0) cbd[(size_t)(b * NHEAD + h) * S_LEN + s] = red[0];
}

// ---------------------------------------------------------------------------
// Flash attention: grid (S/64, B*H), 128 threads = 4 waves, 16 rows/wave.
// QK^T and PV both on v_wmma_f32_16x16x32_bf16; K/V tiles double-buffered via
// async DMA; online softmax in f32 with lane-local running sums.
// ---------------------------------------------------------------------------
__global__ void __launch_bounds__(128, 1)
k_attn(const bf16_t* __restrict__ qp, const bf16_t* __restrict__ kk,
       const bf16_t* __restrict__ vv, const float* __restrict__ cbd,
       bf16_t* __restrict__ avb) {
  __shared__ __align__(16) bf16_t Ks[2][32][64];
  __shared__ __align__(16) bf16_t Vs[2][32][64];
  __shared__ __align__(16) bf16_t Ps[4][16][32];

  const int tid = threadIdx.x, l = tid & 31, w = tid >> 5;
  const int bh = blockIdx.y;
  const int b = bh >> 4, h = bh & 15;
  const int i0 = blockIdx.x * 64 + w * 16;

  const bf16_t* Qb = qp + (size_t)bh * S_LEN * DHEAD;
  const bf16_t* Kb = kk + (size_t)bh * S_LEN * DHEAD;
  const bf16_t* Vb = vv + (size_t)bh * S_LEN * DHEAD;
  const float*  cp = cbd + (size_t)bh * S_LEN;

  // Q' A-fragments for this wave's 16 rows (Dh=64 -> two K=32 frags)
  const int mrow = i0 + (l & 15);
  const int kh = (l >> 4) * 16;
  v16bf a1, a2;
#pragma unroll
  for (int p = 0; p < 8; ++p) {
    int kb = kh + p * 2;
    a1[2 * p]     = Qb[(size_t)mrow * DHEAD + kb];
    a1[2 * p + 1] = Qb[(size_t)mrow * DHEAD + kb + 1];
    a2[2 * p]     = Qb[(size_t)mrow * DHEAD + 32 + kb];
    a2[2 * p + 1] = Qb[(size_t)mrow * DHEAD + 32 + kb + 1];
  }

  const int half8 = (l & 16) ? 8 : 0;
  float cc[8], ccn[8];
#pragma unroll
  for (int p = 0; p < 8; ++p) {
    int i = i0 + half8 + p;
    cc[p] = cp[i];
    int in_ = i + 1; if (in_ > S_LEN - 1) in_ = S_LEN - 1;
    ccn[p] = cp[in_];
  }

  float mrowv[8], srowv[8];   // srowv is LANE-LOCAL partial sum
#pragma unroll
  for (int p = 0; p < 8; ++p) { mrowv[p] = -1e30f; srowv[p] = 0.f; }
  v8f o[4] = {};
  const float scale = 0.125f;  // 1/sqrt(64)

  const int jr = tid >> 2, cg = (tid & 3) * 16;
  const int kbh = (l & 16) ? 16 : 0;

  auto load_kv = [&](int jt, int buf) {
    async_copy_b128(&Ks[buf][jr][cg],     Kb + (size_t)(jt + jr) * DHEAD + cg);
    async_copy_b128(&Ks[buf][jr][cg + 8], Kb + (size_t)(jt + jr) * DHEAD + cg + 8);
    async_copy_b128(&Vs[buf][jr][cg],     Vb + (size_t)(jt + jr) * DHEAD + cg);
    async_copy_b128(&Vs[buf][jr][cg + 8], Vb + (size_t)(jt + jr) * DHEAD + cg + 8);
  };

  load_kv(0, 0);
  wait_async();
  __syncthreads();

  const int njt = S_LEN / 32;
  for (int ks = 0; ks < njt; ++ks) {
    const int jt = ks * 32;
    const int buf = ks & 1;
    if (ks + 1 < njt) load_kv(jt + 32, buf ^ 1);   // overlap DMA with compute

    // S-tile = Q' K^T for two 16-col subtiles, + rel-shifted BD, * scale
    v8f st[2];
#pragma unroll
    for (int t = 0; t < 2; ++t) {
      v16bf b1, b2;
      int jloc = t * 16 + (l & 15);
#pragma unroll
      for (int p = 0; p < 8; ++p) {
        b1[2 * p]     = Ks[buf][jloc][kbh + 2 * p];
        b1[2 * p + 1] = Ks[buf][jloc][kbh + 2 * p + 1];
        b2[2 * p]     = Ks[buf][jloc][32 + kbh + 2 * p];
        b2[2 * p + 1] = Ks[buf][jloc][32 + kbh + 2 * p + 1];
      }
      v8f z = {};
      v8f s = __builtin_amdgcn_wmma_f32_16x16x32_bf16(false, a1, false, b1,
                                                      (short)0, z, false, false);
      s = __builtin_amdgcn_wmma_f32_16x16x32_bf16(false, a2, false, b2,
                                                  (short)0, s, false, false);
#pragma unroll
      for (int p = 0; p < 8; ++p) {
        int i = i0 + half8 + p;
        int j = jt + t * 16 + (l & 15);
        float bd = (j <= i) ? cc[p] : ((j == i + 1) ? 0.0f : ccn[p]);
        s[p] = (s[p] + bd) * scale;
      }
      st[t] = s;
    }

    // Online softmax: one VALU max-reduction per row; sum kept lane-local.
#pragma unroll
    for (int p = 0; p < 8; ++p) {
      float v0 = st[0][p], v1 = st[1][p];
      float lm = redmax16(fmaxf(v0, v1));
      float mn = fmaxf(mrowv[p], lm);
      float alpha = __expf(mrowv[p] - mn);
      mrowv[p] = mn;
      float p0 = __expf(v0 - mn);
      float p1 = __expf(v1 - mn);
      srowv[p] = srowv[p] * alpha + p0 + p1;   // lane-local partial
      Ps[w][half8 + p][l & 15]        = (bf16_t)p0;
      Ps[w][half8 + p][16 + (l & 15)] = (bf16_t)p1;
#pragma unroll
      for (int nc = 0; nc < 4; ++nc) o[nc][p] *= alpha;
    }
    __builtin_amdgcn_wave_barrier();  // wave-local DS ordering fence

    // Re-read P in A-fragment layout
    v16bf pa;
    {
      int m = l & 15;
      int kh2 = (l & 16) ? 16 : 0;
#pragma unroll
      for (int p = 0; p < 8; ++p) {
        int kb = kh2 + p * 2;
        pa[2 * p]     = Ps[w][m][kb];
        pa[2 * p + 1] = Ps[w][m][kb + 1];
      }
    }
    // O += P @ V  (K=32 j-step)
#pragma unroll
    for (int nc = 0; nc < 4; ++nc) {
      v16bf vb;
      int ncol = nc * 16 + (l & 15);
#pragma unroll
      for (int p = 0; p < 8; ++p) {
        vb[2 * p]     = Vs[buf][kbh + 2 * p][ncol];
        vb[2 * p + 1] = Vs[buf][kbh + 2 * p + 1][ncol];
      }
      o[nc] = __builtin_amdgcn_wmma_f32_16x16x32_bf16(false, pa, false, vb,
                                                      (short)0, o[nc], false, false);
    }
    wait_async();      // next K/V tiles have landed in buf^1
    __syncthreads();   // everyone done reading buf / writing buf^1
  }

  // Final cross-lane sum of the lane-local partials, normalize, store.
  float rinv[8];
#pragma unroll
  for (int p = 0; p < 8; ++p) rinv[p] = 1.0f / redsum16(srowv[p]);
#pragma unroll
  for (int nc = 0; nc < 4; ++nc)
#pragma unroll
    for (int p = 0; p < 8; ++p) {
      int i = i0 + half8 + p;
      int col = h * DHEAD + nc * 16 + (l & 15);
      avb[((size_t)i * BATCH + b) * DMODEL + col] = (bf16_t)(o[nc][p] * rinv[p]);
    }
}

// ---------------------------------------------------------------------------
extern "C" void kernel_launch(void* const* d_in, const int* in_sizes, int n_in,
                              void* d_out, int out_size, void* d_ws, size_t ws_size,
                              hipStream_t stream) {
  const float* x    = (const float*)d_in[0];
  const float* r    = (const float*)d_in[1];
  const float* rw   = (const float*)d_in[2];
  const float* rr   = (const float*)d_in[3];
  const float* ln_g = (const float*)d_in[4];
  const float* ln_b = (const float*)d_in[5];
  const float* Wqkv = (const float*)d_in[6];
  const float* Wr   = (const float*)d_in[7];
  const float* Wo   = (const float*)d_in[8];
  float* out = (float*)d_out;

  // workspace carve-up (~118 MB total)
  char* p = (char*)d_ws;
  auto alloc = [&](size_t bytes) -> void* {
    void* q = (void*)p;
    p += (bytes + 255) & ~(size_t)255;
    return q;
  };
  float*  xn    = (float*)alloc((size_t)MROWS * DMODEL * 4);
  bf16_t* xnb   = (bf16_t*)alloc((size_t)MROWS * DMODEL * 2);
  bf16_t* wqkvb = (bf16_t*)alloc((size_t)DMODEL * N3 * 2);
  bf16_t* wob   = (bf16_t*)alloc((size_t)DMODEL * DMODEL * 2);
  float*  qkv   = (float*)alloc((size_t)MROWS * N3 * 4);
  bf16_t* qp    = (bf16_t*)alloc((size_t)BATCH * NHEAD * S_LEN * DHEAD * 2);
  bf16_t* kk    = (bf16_t*)alloc((size_t)BATCH * NHEAD * S_LEN * DHEAD * 2);
  bf16_t* vv    = (bf16_t*)alloc((size_t)BATCH * NHEAD * S_LEN * DHEAD * 2);
  float*  cbd   = (float*)alloc((size_t)BATCH * NHEAD * S_LEN * 4);
  float*  rvec  = (float*)alloc((size_t)DMODEL * 4);
  bf16_t* avb   = (bf16_t*)alloc((size_t)MROWS * DMODEL * 2);
  (void)ws_size; (void)in_sizes; (void)n_in; (void)out_size;

  // 1. weight conversions + LayerNorm + rvec GEMV
  k_cvt<<<(DMODEL * N3 + 255) / 256, 256, 0, stream>>>(Wqkv, wqkvb, DMODEL * N3);
  k_cvt<<<(DMODEL * DMODEL + 255) / 256, 256, 0, stream>>>(Wo, wob, DMODEL * DMODEL);
  k_ln<<<MROWS, 256, 0, stream>>>(x, ln_g, ln_b, xn, xnb);
  k_rvec<<<(DMODEL + 255) / 256, 256, 0, stream>>>(r, Wr, rvec);

  // 2. QKV projection (WMMA, 128x64 block tiles, double-buffered)
  k_gemm<0><<<dim3(N3 / 64, MROWS / 128), 128, 0, stream>>>(
      xnb, wqkvb, nullptr, qkv, MROWS, N3, DMODEL);

  // 3. head split + biases + collapsed BD GEMV
  k_prep<<<MROWS * NHEAD, 64, 0, stream>>>(qkv, rw, rr, rvec, qp, kk, vv, cbd);

  // 4. flash attention (WMMA QK^T + PV, double-buffered K/V)
  k_attn<<<dim3(S_LEN / 64, BATCH * NHEAD), 128, 0, stream>>>(qp, kk, vv, cbd, avb);

  // 5. output projection + residual (WMMA)
  k_gemm<1><<<dim3(DMODEL / 64, MROWS / 128), 128, 0, stream>>>(
      avb, wob, xn, out, MROWS, DMODEL, DMODEL);
}
